// MultiHeadAttention_58677843198063
// MI455X (gfx1250) — matrix operand order repaired and checked by reading, not compiled, not run on hardware
//
#include <hip/hip_runtime.h>

// Problem constants (from reference)
#define D_MODEL 1024
#define NUM_HEADS 16
#define D_K 64
#define SEQ 2048
#define BATCH 2
#define M_TOTAL (BATCH * SEQ) // 4096

// ---- CDNA5 WMMA types -------------------------------------------------------
typedef __attribute__((ext_vector_type(16))) __bf16 v16bf;
typedef __attribute__((ext_vector_type(8)))  float  v8f;

struct alignas(16) U128 { unsigned int u[4]; };     // POD 16B chunk
struct alignas(16) F128 { float f[4]; };            // POD 16B fp32 chunk

union Frag16 {            // one WMMA 16-bit operand fragment (8 VGPRs / lane)
  v16bf v;
  U128 q[2];
  unsigned short s[16];
};

union Pack8 { U128 q; unsigned short s[8]; };

__device__ __forceinline__ unsigned short f2bfu(float f) {
  union { float f; unsigned int u; } c; c.f = f;
  unsigned int u = c.u;
  u += 0x7fffu + ((u >> 16) & 1u);   // round-to-nearest-even
  return (unsigned short)(u >> 16);
}

__device__ __forceinline__ v8f vzero8() {
  v8f z;
#pragma unroll
  for (int i = 0; i < 8; ++i) z[i] = 0.0f;
  return z;
}

// ---- Tensor Data Mover: 2D tile global->LDS (bf16, with LDS row padding) ----
#if __has_builtin(__builtin_amdgcn_tensor_load_to_lds)
#define HAVE_TDM 1
typedef __attribute__((ext_vector_type(4))) unsigned int v4u_;
typedef __attribute__((ext_vector_type(8))) int v8i_;
typedef __attribute__((ext_vector_type(4))) int v4i_;

// d0flags: data_size/pad_enable/pad_interval/pad_amount bits of D# group1 dword0
__device__ __forceinline__ void tdm_load_2d(unsigned lds_addr, const void* gptr,
                                            unsigned d0flags, unsigned tile0,
                                            unsigned tile1,
                                            unsigned long long stride0) {
  unsigned long long ga = (unsigned long long)(size_t)gptr;
  v4u_ g0;
  g0[0] = 1u;                                     // count=1, user descriptor
  g0[1] = lds_addr;                               // LDS byte address
  g0[2] = (unsigned)(ga & 0xFFFFFFFFu);           // global_addr[31:0]
  g0[3] = (unsigned)((ga >> 32) & 0x1FFFFFFu) | (2u << 30); // addr[56:32]|type=2
  const unsigned TD0 = 1u << 20, TD1 = 1u << 20;  // huge tensor dims: no OOB clip
  v8i_ g1;
  g1[0] = (int)d0flags;                           // mask=0,data_size,pad cfg
  g1[1] = (int)((TD0 & 0xFFFFu) << 16);           // tensor_dim0[15:0] @ [63:48]
  g1[2] = (int)((TD0 >> 16) | ((TD1 & 0xFFFFu) << 16));
  g1[3] = (int)((TD1 >> 16) | (tile0 << 16));     // tile_dim0 @ [127:112]
  g1[4] = (int)tile1;                             // tile_dim1; tile_dim2=0
  g1[5] = (int)(unsigned)(stride0 & 0xFFFFFFFFull);       // dim0 stride lo
  g1[6] = (int)(unsigned)((stride0 >> 32) & 0xFFFFull);   // dim0 stride hi
  g1[7] = 0;                                      // dim1 stride (unused, 2D)
  v4i_ z4 = {0, 0, 0, 0};
#if defined(__clang_major__) && (__clang_major__ >= 23)
  v8i_ z8 = {0, 0, 0, 0, 0, 0, 0, 0};
  __builtin_amdgcn_tensor_load_to_lds(g0, g1, z4, z4, z8, 0);
#else
  __builtin_amdgcn_tensor_load_to_lds(g0, g1, z4, z4, 0);
#endif
}
#else
#define HAVE_TDM 0
#endif

// ---- GEMM: Out[m][n] = sum_k A[m][k] * W[n][k] + bias[n] --------------------
// A: fp32 or bf16 (row-major, K=1024 contiguous). W: fp32 row-major [N][K].
// OUT_MODE: 0 = fp32 [m][n], 1 = bf16 [m][n], 2 = bf16 V^T head-major
//           layout [b][h][dk][seq] (packed 16B stores along seq).
template<bool A_BF16, int OUT_MODE>
__global__ __launch_bounds__(256)
void gemm_bias_wmma(const void* __restrict__ Ap,
                    const float* __restrict__ Wp,
                    const float* __restrict__ bias,
                    void* __restrict__ Outp)
{
  constexpr int K   = D_MODEL;
  constexpr int LDT = 40; // LDS row stride (ushorts): 80B, 16B-aligned, skewed
  __shared__ __align__(16) unsigned short sA[128 * LDT];
  __shared__ __align__(16) unsigned short sB[128 * LDT];

  const int tid    = threadIdx.x;
  const int lane   = tid & 31;
  const int wave   = tid >> 5;
  const int m_off  = (wave & 3) * 32;
  const int n_off  = (wave >> 2) * 64;
  const int blockM = blockIdx.x * 128;
  const int blockN = blockIdx.y * 128;

  v8f acc[2][4];
#pragma unroll
  for (int mi = 0; mi < 2; ++mi)
#pragma unroll
    for (int ni = 0; ni < 4; ++ni) acc[mi][ni] = vzero8();

  const int srow = tid >> 1;        // 0..127
  const int scol = (tid & 1) * 16;  // 0 or 16

  for (int kk = 0; kk < K; kk += 32) {
    // ---- stage A tile (128x32) as bf16 in LDS ----
    {
      Frag16 ua;
      if constexpr (A_BF16) {
        const U128* src = (const U128*)((const unsigned short*)Ap +
                          (size_t)(blockM + srow) * K + kk + scol);
        ua.q[0] = src[0]; ua.q[1] = src[1];
        if (kk + 32 < K) __builtin_prefetch((const char*)src + 64, 0, 1);
      } else {
        const float* src = (const float*)Ap + (size_t)(blockM + srow) * K + kk + scol;
        float tmp[16] __attribute__((aligned(16)));
#pragma unroll
        for (int c = 0; c < 4; ++c) *(F128*)&tmp[c * 4] = ((const F128*)src)[c];
#pragma unroll
        for (int e = 0; e < 16; ++e) ua.s[e] = f2bfu(tmp[e]);
        if (kk + 32 < K) __builtin_prefetch((const char*)(src + 32), 0, 1);
      }
      *(U128*)&sA[srow * LDT + scol]     = ua.q[0];
      *(U128*)&sA[srow * LDT + scol + 8] = ua.q[1];
    }
    // ---- stage W tile (128 N-rows x 32 K) as bf16 in LDS ----
    {
      const float* src = Wp + (size_t)(blockN + srow) * K + kk + scol;
      float tmp[16] __attribute__((aligned(16)));
#pragma unroll
      for (int c = 0; c < 4; ++c) *(F128*)&tmp[c * 4] = ((const F128*)src)[c];
      Frag16 ub;
#pragma unroll
      for (int e = 0; e < 16; ++e) ub.s[e] = f2bfu(tmp[e]);
      *(U128*)&sB[srow * LDT + scol]     = ub.q[0];
      *(U128*)&sB[srow * LDT + scol + 8] = ub.q[1];
    }
    __syncthreads();

    // ---- fragments + 8 WMMAs ----
    Frag16 afr[2], bfr[4];
    const int am  = lane & 15;
    const int akb = (lane >> 4) * 8;
#pragma unroll
    for (int mi = 0; mi < 2; ++mi) {
      const unsigned short* p = &sA[(m_off + mi * 16 + am) * LDT + akb];
      afr[mi].q[0] = *(const U128*)p;
      afr[mi].q[1] = *(const U128*)(p + 16);
    }
    const int bn  = lane & 15;
    const int bkb = (lane >> 4) * 8;
#pragma unroll
    for (int ni = 0; ni < 4; ++ni) {
      const unsigned short* p = &sB[(n_off + ni * 16 + bn) * LDT + bkb];
      bfr[ni].q[0] = *(const U128*)p;
      bfr[ni].q[1] = *(const U128*)(p + 16);
    }
#pragma unroll
    for (int mi = 0; mi < 2; ++mi)
#pragma unroll
      for (int ni = 0; ni < 4; ++ni)
        acc[mi][ni] = __builtin_amdgcn_wmma_f32_16x16x32_bf16(
            false, afr[mi].v, false, bfr[ni].v, (short)0, acc[mi][ni], false, false);
    __syncthreads();
  }

  // ---- epilogue: C layout = VGPR j -> row j (lanes 0-15) / j+8 (16-31) ----
#pragma unroll
  for (int mi = 0; mi < 2; ++mi) {
#pragma unroll
    for (int ni = 0; ni < 4; ++ni) {
      const int n  = blockN + n_off + ni * 16 + (lane & 15);
      const float bb = bias[n];
      if constexpr (OUT_MODE == 2) {
        // V^T head-major: [b][h][dk][seq]; lane's 8 rows = 8 consecutive seq
        const int m0 = blockM + m_off + mi * 16 + ((lane >> 4) * 8);
        const int bt = m0 >> 11;          // batch (SEQ = 2048)
        const int s0 = m0 & (SEQ - 1);
        Pack8 pk;
#pragma unroll
        for (int j = 0; j < 8; ++j) pk.s[j] = f2bfu(acc[mi][ni][j] + bb);
        unsigned short* dst = (unsigned short*)Outp +
            (((size_t)bt * NUM_HEADS + (n >> 6)) * D_K + (n & 63)) * SEQ + s0;
        *(U128*)dst = pk.q;
      } else {
#pragma unroll
        for (int j = 0; j < 8; ++j) {
          const int m = blockM + m_off + mi * 16 + j + ((lane >> 4) * 8);
          const float val = acc[mi][ni][j] + bb;
          if constexpr (OUT_MODE == 1)
            ((unsigned short*)Outp)[(size_t)m * D_MODEL + n] = f2bfu(val);
          else
            ((float*)Outp)[(size_t)m * D_MODEL + n] = val;
        }
      }
    }
  }
}

// ---- Flash attention per (b, h): online softmax, WMMA for QK^T and PV -------
// Block: 128 threads = 4 waves, 64 query rows (16 per wave). Key tile = 32.
// K tile staged row-major [key][dk]; V^T staged row-major [dk][key] from the
// head-major V^T global layout. Staging via TDM (tensor_load_to_lds) when the
// toolchain exposes it, manual vector copies otherwise.
__global__ __launch_bounds__(128)
void flash_attn_wmma(const unsigned short* __restrict__ Qb,
                     const unsigned short* __restrict__ Kb,
                     const unsigned short* __restrict__ Vt,
                     unsigned short* __restrict__ Ob)
{
  constexpr int LDK = 72; // 64 elems (128B) + 16B pad per key row
  constexpr int LDV = 40; // 32 elems (64B) + 16B pad per dk row
  constexpr int LDP = 40;
  __shared__ __align__(16) unsigned short sK [32 * LDK];
  __shared__ __align__(16) unsigned short sVt[64 * LDV];
  __shared__ __align__(16) unsigned short sP [4 * 16 * LDP];

  const int tid  = threadIdx.x;
  const int lane = tid & 31;
  const int wid  = tid >> 5;
  const int q0   = blockIdx.x * 64;
  const int h    = blockIdx.y;
  const int b    = blockIdx.z;

  const size_t headoff = (size_t)h * D_K;
  const size_t basebs  = (size_t)b * SEQ;
  const unsigned short* Vhead =
      Vt + ((size_t)b * NUM_HEADS + h) * D_K * SEQ;  // [dk][seq]

  // Persistent Q fragments: A-matrix 16x64 split into two K=32 fragments
  Frag16 qf[2];
  {
    const int m  = lane & 15;
    const int kb = (lane >> 4) * 8;
    const unsigned short* qrow =
        Qb + (basebs + q0 + wid * 16 + m) * D_MODEL + headoff;
#pragma unroll
    for (int kh = 0; kh < 2; ++kh) {
      qf[kh].q[0] = *(const U128*)(qrow + kh * 32 + kb);
      qf[kh].q[1] = *(const U128*)(qrow + kh * 32 + kb + 16);
    }
  }

#if HAVE_TDM
  const unsigned sK_lds  = (unsigned)(size_t)&sK[0];
  const unsigned sVt_lds = (unsigned)(size_t)&sVt[0];
#endif

  float ms[8], ls[8];
  v8f acc[4];
#pragma unroll
  for (int j = 0; j < 8; ++j) { ms[j] = -3.0e38f; ls[j] = 0.0f; }
#pragma unroll
  for (int dt = 0; dt < 4; ++dt) acc[dt] = vzero8();

  for (int kb0 = 0; kb0 < SEQ; kb0 += 32) {
#if HAVE_TDM
    // ---- TDM: DMA K tile (32x64) and V^T tile (64x32) into padded LDS ----
    if (wid == 0) {
      // data_size=2B (1<<16), pad_enable (1<<20),
      // K: pad_interval 128B (4<<22), pad 16B (3<<25) -> row stride 144B
      tdm_load_2d(sK_lds, Kb + (basebs + kb0) * D_MODEL + headoff,
                  0x07110000u, 64u, 32u, (unsigned long long)D_MODEL);
      // V: pad_interval 64B (3<<22), pad 16B (3<<25) -> row stride 80B
      tdm_load_2d(sVt_lds, Vhead + kb0,
                  0x06D10000u, 32u, 64u, (unsigned long long)SEQ);
      __builtin_amdgcn_s_wait_tensorcnt(0);
    }
    __syncthreads();
#else
    // ---- manual staging fallback (both tiles row-major, no transpose) ----
    {
      const int row = tid >> 2;        // key 0..31
      const int ch  = (tid & 3) * 16;
      const unsigned short* ksrc =
          Kb + (basebs + kb0 + row) * D_MODEL + headoff + ch;
      *(U128*)&sK[row * LDK + ch]     = *(const U128*)(ksrc);
      *(U128*)&sK[row * LDK + ch + 8] = *(const U128*)(ksrc + 8);

      const int vrow = tid >> 1;       // dk 0..63
      const int vch  = (tid & 1) * 16; // key chunk
      const unsigned short* vsrc = Vhead + (size_t)vrow * SEQ + kb0 + vch;
      *(U128*)&sVt[vrow * LDV + vch]     = *(const U128*)(vsrc);
      *(U128*)&sVt[vrow * LDV + vch + 8] = *(const U128*)(vsrc + 8);
    }
    __syncthreads();
#endif

    // ---- scores S = (Q K^T) / sqrt(dk): two 16x16 tiles over 32 keys ----
    v8f sc[2];
#pragma unroll
    for (int nt = 0; nt < 2; ++nt) {
      v8f z = vzero8();
#pragma unroll
      for (int kh = 0; kh < 2; ++kh) {
        Frag16 kfr; // B[k=dk][n=key] == K row-major read
        const unsigned short* p =
            &sK[(nt * 16 + (lane & 15)) * LDK + kh * 32 + (lane >> 4) * 8];
        kfr.q[0] = *(const U128*)p;
        kfr.q[1] = *(const U128*)(p + 16);
        z = __builtin_amdgcn_wmma_f32_16x16x32_bf16(
            false, qf[kh].v, false, kfr.v, (short)0, z, false, false);
      }
      sc[nt] = z * 0.125f;  // 1/sqrt(64)
    }

    // ---- online softmax; rows live in 16-lane groups (shfl_xor <= 8) ----
#pragma unroll
    for (int j = 0; j < 8; ++j) {
      float t = fmaxf(sc[0][j], sc[1][j]);
      t = fmaxf(t, __shfl_xor(t, 1, 32));
      t = fmaxf(t, __shfl_xor(t, 2, 32));
      t = fmaxf(t, __shfl_xor(t, 4, 32));
      t = fmaxf(t, __shfl_xor(t, 8, 32));
      const float mnew = fmaxf(ms[j], t);
      const float resc = __expf(ms[j] - mnew);
      const float p0 = __expf(sc[0][j] - mnew);
      const float p1 = __expf(sc[1][j] - mnew);

      const int prow = j + ((lane >> 4) * 8);
      const int pcol = lane & 15;
      unsigned short* pp = &sP[(wid * 16 + prow) * LDP + pcol];
      pp[0]  = f2bfu(p0);
      pp[16] = f2bfu(p1);

      float rs = p0 + p1;
      rs += __shfl_xor(rs, 1, 32);
      rs += __shfl_xor(rs, 2, 32);
      rs += __shfl_xor(rs, 4, 32);
      rs += __shfl_xor(rs, 8, 32);

      ls[j] = ls[j] * resc + rs;
      ms[j] = mnew;
#pragma unroll
      for (int dt = 0; dt < 4; ++dt) acc[dt][j] *= resc;
    }

    // wave-local LDS RAW: wait on the split DS counter before re-reading P
    asm volatile("s_wait_dscnt 0" ::: "memory");

    // ---- P as A-fragment (16 q x 32 keys), then PV accumulation ----
    Frag16 pf;
    {
      const int m  = lane & 15;
      const int kb = (lane >> 4) * 8;
      const unsigned short* p = &sP[(wid * 16 + m) * LDP + kb];
      pf.q[0] = *(const U128*)p;
      pf.q[1] = *(const U128*)(p + 16);
    }
#pragma unroll
    for (int dt = 0; dt < 4; ++dt) {
      Frag16 vf; // B[k=key][n=dk] from V^T tile
      const unsigned short* p =
          &sVt[(dt * 16 + (lane & 15)) * LDV + (lane >> 4) * 8];
      vf.q[0] = *(const U128*)p;
      vf.q[1] = *(const U128*)(p + 16);
      acc[dt] = __builtin_amdgcn_wmma_f32_16x16x32_bf16(
          false, pf.v, false, vf.v, (short)0, acc[dt], false, false);
    }
    __syncthreads();
  }

  // ---- normalize + store merged-head bf16 output ----
#pragma unroll
  for (int j = 0; j < 8; ++j) {
    const float inv = 1.0f / ls[j];
    const int m = q0 + wid * 16 + j + ((lane >> 4) * 8);
#pragma unroll
    for (int dt = 0; dt < 4; ++dt) {
      const int d = dt * 16 + (lane & 15);
      Ob[(basebs + m) * D_MODEL + headoff + d] = f2bfu(acc[dt][j] * inv);
    }
  }
}

// ---- host-side launch -------------------------------------------------------
extern "C" void kernel_launch(void* const* d_in, const int* in_sizes, int n_in,
                              void* d_out, int out_size, void* d_ws, size_t ws_size,
                              hipStream_t stream) {
  const float* q   = (const float*)d_in[0];
  const float* k   = (const float*)d_in[1];
  const float* v   = (const float*)d_in[2];
  const float* w_q = (const float*)d_in[3];
  const float* b_q = (const float*)d_in[4];
  const float* w_k = (const float*)d_in[5];
  const float* b_k = (const float*)d_in[6];
  const float* w_v = (const float*)d_in[7];
  const float* b_v = (const float*)d_in[8];
  const float* w_o = (const float*)d_in[9];
  const float* b_o = (const float*)d_in[10];

  const size_t E = (size_t)M_TOTAL * D_MODEL;  // 4Mi elements per tensor
  unsigned short* Qb  = (unsigned short*)d_ws; // bf16 workspaces
  unsigned short* Kb  = Qb + E;
  unsigned short* Vtg = Kb + E;                // V^T head-major [b][h][dk][seq]
  unsigned short* Ab  = Vtg + E;

  dim3 gg(M_TOTAL / 128, D_MODEL / 128, 1);    // (32, 8)
  gemm_bias_wmma<false, 1><<<gg, 256, 0, stream>>>(q, w_q, b_q, Qb);
  gemm_bias_wmma<false, 1><<<gg, 256, 0, stream>>>(k, w_k, b_k, Kb);
  gemm_bias_wmma<false, 2><<<gg, 256, 0, stream>>>(v, w_v, b_v, Vtg);

  dim3 ga(SEQ / 64, NUM_HEADS, BATCH);         // (32, 16, 2)
  flash_attn_wmma<<<ga, 128, 0, stream>>>(Qb, Kb, Vtg, Ab);

  gemm_bias_wmma<true, 0><<<gg, 256, 0, stream>>>(Ab, w_o, b_o, (float*)d_out);
}
